// MultiContextGating_22101901705856
// MI455X (gfx1250) — matrix operand order repaired and checked by reading, not compiled, not run on hardware
//
#include <hip/hip_runtime.h>
#include <float.h>

typedef __attribute__((ext_vector_type(16))) __bf16 v16bf;
typedef __attribute__((ext_vector_type(8)))  float  v8f;

#define HDIM 512
#define ADIM 128
#define NCTX 4
#define NTHREADS 512   // 16 wave32 waves

// Convert W_feat (f32, [4][o=512][h=512]) into bf16 B-fragment-swizzled layout:
// index = ((mat*256k) + ((kt*32 + nt)*32 + lane)*16 + slot), with
//   n = nt*16 + (lane&15),  k = kt*32 + (slot&7) + 8*(((slot>>3)<<1)|(lane>>4))   (lane=N col,
//   lanes 0-15 hold K 0-7,16-23 in slots, lanes 16-31 hold K 8-15,24-31; matches A order)
__global__ __launch_bounds__(256) void swizzle_wfeat_kernel(const float* __restrict__ Wf,
                                                            __bf16* __restrict__ out) {
  int i = blockIdx.x * blockDim.x + threadIdx.x;   // 0 .. 4*512*512-1
  int mat  = i >> 18;
  int j    = i & 0x3FFFF;
  int s    = j & 15;
  int lane = (j >> 4) & 31;
  int nt   = (j >> 9) & 31;
  int kt   = j >> 14;
  int n = nt * 16 + (lane & 15);
  int k = kt * 32 + (s & 7) + ((((s >> 3) << 1) | (lane >> 4)) << 3);
  out[i] = (__bf16)Wf[((size_t)mat * HDIM + n) * HDIM + k];
}

// One workgroup per batch element b. prev_h master (f32) lives in d_out (L2-resident);
// bf16 A-swizzled copy ping-pongs in LDS.
__global__ __launch_bounds__(NTHREADS, 1) void mcg_kernel(
    const float* __restrict__ hidden,
    const unsigned char* __restrict__ avail,
    const __bf16* __restrict__ WfB,
    const float* __restrict__ bfeat_all,
    const float* __restrict__ Wctx,
    const float* __restrict__ bctx_all,
    float* __restrict__ out)
{
  __shared__ __bf16 hbuf[2][ADIM * HDIM];   // 2 x 128 KiB, A-fragment swizzled bf16
  __shared__ float  ctxS[HDIM];
  __shared__ float  ctxNew[HDIM];
  __shared__ float  pcS[HDIM];
  __shared__ int    maskS[ADIM];

  const int b    = blockIdx.x;
  const int t    = threadIdx.x;
  const int lane = t & 31;
  const int wave = t >> 5;                  // 0..15

  const float* hsrc = hidden + (size_t)b * ADIM * HDIM;
  float*       odst = out    + (size_t)b * ADIM * HDIM;

  for (int o = t; o < HDIM; o += NTHREADS) {
    ctxS[o] = 1.0f; pcS[o] = 1.0f; ctxNew[o] = -FLT_MAX;
  }
  if (t < ADIM) maskS[t] = (int)avail[b * ADIM + t];

  // Build initial A-swizzled bf16 prev_h from hidden.
  // A layout (16x32 bf16): lane: m=lane&15, hi=lane>>4; slot s -> k = (s&7) + 8*(((s>>3)<<1)|hi)
  for (int i = t; i < ADIM * HDIM; i += NTHREADS) {
    int s  = i & 15;
    int ln = (i >> 4) & 31;
    int kt = (i >> 9) & 15;
    int mt = i >> 13;
    int a  = mt * 16 + (ln & 15);
    int k  = kt * 32 + (s & 7) + ((((s >> 3) << 1) | (ln >> 4)) << 3);
    hbuf[0][i] = (__bf16)hsrc[a * HDIM + k];
  }
  __syncthreads();

  const int mt     = wave & 7;              // M tile (a dimension)
  const int ntbase = (wave >> 3) << 4;      // N tile base (o dimension)
  int cur = 0;

  for (int idx = 0; idx < NCTX; ++idx) {
    const float inv_i = 1.0f / (float)(idx + 1);
    const float* osrc  = (idx == 0) ? hsrc : odst;
    const float* bfeat = bfeat_all + idx * HDIM;
    const __bf16* Wmat = WfB + (size_t)idx * HDIM * HDIM;
    const int nxt = cur ^ 1;

    for (int g = 0; g < 4; ++g) {           // 4 groups of 4 N tiles -> 16 N tiles per wave
      v8f acc[4];
      #pragma unroll
      for (int j2 = 0; j2 < 4; ++j2)
        acc[j2] = (v8f){0.f,0.f,0.f,0.f,0.f,0.f,0.f,0.f};
      const int nt0 = ntbase + g * 4;

      for (int kt = 0; kt < 16; ++kt) {     // K = 512 in steps of 32
        v16bf afrag = *(const v16bf*)&hbuf[cur][((mt * 16 + kt) * 32 + lane) * 16];
        #pragma unroll
        for (int j2 = 0; j2 < 4; ++j2) {
          v16bf bfrag = *(const v16bf*)
              &Wmat[(size_t)(((kt * 32 + (nt0 + j2)) * 32 + lane) * 16)];
          acc[j2] = __builtin_amdgcn_wmma_f32_16x16x32_bf16(
              false, afrag, false, bfrag, (short)0, acc[j2], false, false);
        }
      }

      // Epilogue for these 4 tiles: gate by ctx, masked max, running-mean update,
      // and scatter bf16 prev_h into the ping-pong buffer for the next iteration.
      const int hi = lane >> 4;
      #pragma unroll
      for (int j2 = 0; j2 < 4; ++j2) {
        const int n0  = (nt0 + j2) * 16 + (lane & 15);  // o column (C layout: lane&15 = N)
        const float cv = ctxS[n0];
        const float bv = bfeat[n0];
        // target A-fragment coords for k = n0 (next iteration's K index)
        const int kt2 = n0 >> 5;
        const int kin = n0 & 31;
        const int blk = kin >> 3;
        const int s2  = (kin & 7) + ((blk >> 1) << 3);
        const int hi2 = blk & 1;
        #pragma unroll
        for (int r = 0; r < 8; ++r) {
          const int a = mt * 16 + hi * 8 + r;          // C layout: VGPR r, lane half -> M
          const float v = (acc[j2][r] + bv) * cv;
          if (maskS[a])
            __hip_atomic_fetch_max(&ctxNew[n0], v, __ATOMIC_RELAXED,
                                   __HIP_MEMORY_SCOPE_WORKGROUP);
          const float nv = osrc[a * HDIM + n0] + v * inv_i;
          odst[a * HDIM + n0] = nv;                    // same lane owns (a,o) every iter
          const int lane2 = (a & 15) + (hi2 << 4);
          hbuf[nxt][(((a >> 4) * 16 + kt2) * 32 + lane2) * 16 + s2] = (__bf16)nv;
        }
      }
    }
    __syncthreads();

    // prev_c update (one o per thread; NTHREADS == HDIM)
    pcS[t] += ctxNew[t] * inv_i;
    __syncthreads();
    ctxNew[t] = -FLT_MAX;
    if (idx < NCTX - 1) {
      // ctx for next iteration: ctx[o] = b_ctx[idx][o] + sum_h pc[h] * W_ctx[idx][o][h]
      const float* wrow = Wctx + ((size_t)idx * HDIM + t) * HDIM;
      float sacc = bctx_all[idx * HDIM + t];
      for (int h2 = 0; h2 < HDIM; h2 += 4) {
        const float4 w4 = *(const float4*)&wrow[h2];
        sacc += pcS[h2] * w4.x + pcS[h2 + 1] * w4.y + pcS[h2 + 2] * w4.z + pcS[h2 + 3] * w4.w;
      }
      ctxS[t] = sacc;
    }
    __syncthreads();
    cur = nxt;
  }
  // final prev_h already resides in d_out
}

extern "C" void kernel_launch(void* const* d_in, const int* in_sizes, int n_in,
                              void* d_out, int out_size, void* d_ws, size_t ws_size,
                              hipStream_t stream) {
  const float*         hidden = (const float*)d_in[0];
  const unsigned char* avail  = (const unsigned char*)d_in[1];
  const float*         Wf     = (const float*)d_in[2];
  const float*         bf     = (const float*)d_in[3];
  const float*         Wc     = (const float*)d_in[4];
  const float*         bc     = (const float*)d_in[5];
  __bf16* wsWf = (__bf16*)d_ws;                    // 4*512*512 bf16 = 2 MiB

  const int totalW = NCTX * HDIM * HDIM;           // 1,048,576
  swizzle_wfeat_kernel<<<totalW / 256, 256, 0, stream>>>(Wf, wsWf);
  mcg_kernel<<<256, NTHREADS, 0, stream>>>(hidden, avail, wsWf, bf, Wc, bc,
                                           (float*)d_out);
}
